// DecoderBlock_69063074120147
// MI455X (gfx1250) — matrix operand order repaired and checked by reading, not compiled, not run on hardware
//
#include <hip/hip_runtime.h>

// ---------------------------------------------------------------------------
// DecoderBlock for MI455X (gfx1250): bf16 WMMA GEMMs + flash-attention.
// B=4, S=1024, D=1024, H=16, DH=64, F=4096, N=B*S=4096.
// All matmuls use v_wmma_f32_16x16x32_bf16 (fp32 accumulate).
// Staging uses gfx1250 async global->LDS copies (ASYNCcnt path) when the
// toolchain exposes them, else register-batched load/store.
// ---------------------------------------------------------------------------

typedef __bf16 bf16_t;
typedef bf16_t v16bf __attribute__((ext_vector_type(16)));
typedef float  v8f   __attribute__((ext_vector_type(8)));

union Frag16 {            // one WMMA 16-bit operand: 8 VGPRs = 32 bytes
    v16bf v;
    uint4 u[2];
};

#define CB 4
#define CS 1024
#define CD 1024
#define CH 16
#define CDH 64
#define CF 4096
#define CN (CB * CS)      // 4096 rows

#if defined(__HIP_DEVICE_COMPILE__) && __has_builtin(__builtin_amdgcn_global_load_async_to_lds_b128)
#define HAVE_ASYNC_LDS 1
typedef int v4i __attribute__((ext_vector_type(4)));
typedef __attribute__((address_space(1))) v4i as1_v4i;   // global
typedef __attribute__((address_space(3))) v4i as3_v4i;   // LDS
#endif

__device__ __forceinline__ void async_g2l_b128(const void* g, void* l) {
#ifdef HAVE_ASYNC_LDS
    __builtin_amdgcn_global_load_async_to_lds_b128((as1_v4i*)g, (as3_v4i*)l, 0, 0);
#else
    (void)g; (void)l;
#endif
}
__device__ __forceinline__ void async_wait_all() {
#ifdef HAVE_ASYNC_LDS
#if __has_builtin(__builtin_amdgcn_s_wait_asynccnt)
    __builtin_amdgcn_s_wait_asynccnt(0);
#else
    asm volatile("s_wait_asynccnt 0" ::: "memory");
#endif
#endif
}

// ---------------------------------------------------------------------------
// fp32 -> bf16 elementwise convert (4 elems/thread)
// ---------------------------------------------------------------------------
__global__ __launch_bounds__(256) void cvt_f32_bf16(const float* __restrict__ in,
                                                    bf16_t* __restrict__ out, int n4) {
    int i = blockIdx.x * 256 + threadIdx.x;
    if (i < n4) {
        float4 v = ((const float4*)in)[i];
        out[i * 4 + 0] = (bf16_t)v.x;
        out[i * 4 + 1] = (bf16_t)v.y;
        out[i * 4 + 2] = (bf16_t)v.z;
        out[i * 4 + 3] = (bf16_t)v.w;
    }
}

// ---------------------------------------------------------------------------
// Batched transpose+convert: in fp32 [G][R][C] -> out bf16 [G][C][R]
// Pre-transposes every weight matrix into the WMMA B-operand [N][K] layout.
// ---------------------------------------------------------------------------
__global__ __launch_bounds__(256) void pack_transpose(const float* __restrict__ in,
                                                      bf16_t* __restrict__ out,
                                                      int G, int R, int C) {
    long long i = (long long)blockIdx.x * 256 + threadIdx.x;
    long long total = (long long)G * R * C;
    if (i >= total) return;
    long long rc = (long long)R * C;
    long long g = i / rc, rem = i - g * rc;
    long long r = rem / C, c = rem - r * C;
    out[g * rc + c * R + r] = (bf16_t)in[i];
}

// ---------------------------------------------------------------------------
// GEMM: out[M][Nc] = A[M][K] (bf16) x Bt[Nc][K]^T (bf16) + bias, opt ReLU.
// Block: 256 threads / 8 waves; block tile 128(M) x 64(N); K-tile 64.
// Wave w owns rows [w*16, w*16+16), all 64 columns (4 x 16x16 accum tiles).
// ---------------------------------------------------------------------------
#define GM 128
#define GN 64
#define GK 64

__global__ __launch_bounds__(256) void gemm_bf16(const bf16_t* __restrict__ A,
                                                 const bf16_t* __restrict__ Bt,
                                                 const float* __restrict__ bias,
                                                 float* __restrict__ outF,
                                                 bf16_t* __restrict__ outB,
                                                 int M, int Nc, int K, int relu) {
    __shared__ bf16_t As[GM * GK];   // 16 KB
    __shared__ bf16_t Bs[GN * GK];   //  8 KB

    const int tid  = threadIdx.x;
    const int lane = tid & 31;
    const int w    = tid >> 5;       // wave 0..7
    const int half = lane >> 4;      // 0 | 1
    const int l16  = lane & 15;
    const int m0   = blockIdx.x * GM;
    const int n0   = blockIdx.y * GN;

    v8f acc[4];
#pragma unroll
    for (int t = 0; t < 4; ++t)
#pragma unroll
        for (int e = 0; e < 8; ++e) acc[t][e] = 0.0f;

    for (int k0 = 0; k0 < K; k0 += GK) {
#ifdef HAVE_ASYNC_LDS
        // gfx1250 async global->LDS copies (ASYNCcnt), no VGPR round-trip
#pragma unroll
        for (int i = 0; i < 4; ++i) {
            int c = tid + i * 256, r = c >> 3, ch = c & 7;
            async_g2l_b128(&A[(size_t)(m0 + r) * K + k0 + ch * 8], &As[r * GK + ch * 8]);
        }
#pragma unroll
        for (int i = 0; i < 2; ++i) {
            int c = tid + i * 256, r = c >> 3, ch = c & 7;
            async_g2l_b128(&Bt[(size_t)(n0 + r) * K + k0 + ch * 8], &Bs[r * GK + ch * 8]);
        }
        async_wait_all();
#else
        // batch all global loads first so they pipeline, then store to LDS
        uint4 ta[4], tb[2];
#pragma unroll
        for (int i = 0; i < 4; ++i) {
            int c = tid + i * 256, r = c >> 3, ch = c & 7;
            ta[i] = *(const uint4*)&A[(size_t)(m0 + r) * K + k0 + ch * 8];
        }
#pragma unroll
        for (int i = 0; i < 2; ++i) {
            int c = tid + i * 256, r = c >> 3, ch = c & 7;
            tb[i] = *(const uint4*)&Bt[(size_t)(n0 + r) * K + k0 + ch * 8];
        }
#pragma unroll
        for (int i = 0; i < 4; ++i) {
            int c = tid + i * 256, r = c >> 3, ch = c & 7;
            *(uint4*)&As[r * GK + ch * 8] = ta[i];
        }
#pragma unroll
        for (int i = 0; i < 2; ++i) {
            int c = tid + i * 256, r = c >> 3, ch = c & 7;
            *(uint4*)&Bs[r * GK + ch * 8] = tb[i];
        }
#endif
        if (k0 + GK < K) {  // gfx1250 global_prefetch for the next A tile
            __builtin_prefetch(&A[(size_t)(m0 + (tid >> 1)) * K + k0 + GK], 0, 1);
        }
        __syncthreads();

#pragma unroll
        for (int kk = 0; kk < GK; kk += 32) {
            // preload A frag + all 4 B frags, then issue WMMAs back-to-back
            Frag16 a, b[4];
            const int arow = w * 16 + l16;
            a.u[0] = *(const uint4*)&As[arow * GK + kk + half * 8];
            a.u[1] = *(const uint4*)&As[arow * GK + kk + half * 8 + 16];
#pragma unroll
            for (int t = 0; t < 4; ++t) {
                const int brow = t * 16 + l16;
                b[t].u[0] = *(const uint4*)&Bs[brow * GK + kk + half * 8];
                b[t].u[1] = *(const uint4*)&Bs[brow * GK + kk + half * 8 + 16];
            }
#pragma unroll
            for (int t = 0; t < 4; ++t)
                acc[t] = __builtin_amdgcn_wmma_f32_16x16x32_bf16(
                    false, a.v, false, b[t].v, (short)0, acc[t], false, false);
        }
        __syncthreads();
    }

    // Epilogue. C layout: VGPR v, lane -> row v + 8*half, col l16.
#pragma unroll
    for (int t = 0; t < 4; ++t) {
        const int col = n0 + t * 16 + l16;
        float bv = bias ? bias[col] : 0.0f;
#pragma unroll
        for (int v = 0; v < 8; ++v) {
            const int row = m0 + w * 16 + v + half * 8;
            float x = acc[t][v] + bv;
            if (relu) x = fmaxf(x, 0.0f);
            if (outF) outF[(size_t)row * Nc + col] = x;
            if (outB) outB[(size_t)row * Nc + col] = (bf16_t)x;
        }
    }
}

// ---------------------------------------------------------------------------
// Flash attention, one (b,h) per blockIdx.y, 128 queries per blockIdx.x.
// Q,K,V bf16 with layout [b*S + s][h*64 + e]; output Ob bf16, same layout.
// 8 waves, each owns 16 query rows; keys streamed in 64-wide tiles.
// ---------------------------------------------------------------------------
__global__ __launch_bounds__(256) void attention_kernel(const bf16_t* __restrict__ Qb,
                                                        const bf16_t* __restrict__ Kb,
                                                        const bf16_t* __restrict__ Vb,
                                                        bf16_t* __restrict__ Ob,
                                                        int causal) {
    __shared__ bf16_t Qs[128 * 64];      // 16 KB
    __shared__ bf16_t Ks[64 * 64];       //  8 KB
    __shared__ bf16_t Vts[64 * 64];      //  8 KB (transposed: [dh][key])
    __shared__ bf16_t Ps[8 * 16 * 64];   // 16 KB (per-wave P strips)

    const int tid  = threadIdx.x;
    const int lane = tid & 31;
    const int w    = tid >> 5;
    const int half = lane >> 4;
    const int l16  = lane & 15;
    const int h    = blockIdx.y % CH;
    const int bb   = blockIdx.y / CH;
    const int qBase = blockIdx.x * 128;
    const size_t rowBase = (size_t)bb * CS;
    const float scale = 0.125f;          // 1/sqrt(64)

    // stage Q tile once
#ifdef HAVE_ASYNC_LDS
#pragma unroll
    for (int i = 0; i < 4; ++i) {
        int c = tid + i * 256, r = c >> 3, ch = c & 7;
        async_g2l_b128(&Qb[(rowBase + qBase + r) * CD + h * 64 + ch * 8],
                       &Qs[r * 64 + ch * 8]);
    }
    async_wait_all();
#else
    {
        uint4 tq[4];
#pragma unroll
        for (int i = 0; i < 4; ++i) {
            int c = tid + i * 256, r = c >> 3, ch = c & 7;
            tq[i] = *(const uint4*)&Qb[(rowBase + qBase + r) * CD + h * 64 + ch * 8];
        }
#pragma unroll
        for (int i = 0; i < 4; ++i) {
            int c = tid + i * 256, r = c >> 3, ch = c & 7;
            *(uint4*)&Qs[r * 64 + ch * 8] = tq[i];
        }
    }
#endif

    v8f oacc[4];
    float mrow[8], lrow[8];
#pragma unroll
    for (int t = 0; t < 4; ++t)
#pragma unroll
        for (int e = 0; e < 8; ++e) oacc[t][e] = 0.0f;
#pragma unroll
    for (int v = 0; v < 8; ++v) { mrow[v] = -1.0e30f; lrow[v] = 0.0f; }

    const int ktEnd = causal ? (qBase + 128) : CS;
    for (int kt = 0; kt < ktEnd; kt += 64) {
        __syncthreads();  // previous iteration's Ks/Vts reads complete
        // stage K tile (row-major == WMMA B-operand layout)
#ifdef HAVE_ASYNC_LDS
#pragma unroll
        for (int i = 0; i < 2; ++i) {
            int c = tid + i * 256, r = c >> 3, ch = c & 7;
            async_g2l_b128(&Kb[(rowBase + kt + r) * CD + h * 64 + ch * 8],
                           &Ks[r * 64 + ch * 8]);
        }
#else
        {
            uint4 tk[2];
#pragma unroll
            for (int i = 0; i < 2; ++i) {
                int c = tid + i * 256, r = c >> 3, ch = c & 7;
                tk[i] = *(const uint4*)&Kb[(rowBase + kt + r) * CD + h * 64 + ch * 8];
            }
#pragma unroll
            for (int i = 0; i < 2; ++i) {
                int c = tid + i * 256, r = c >> 3, ch = c & 7;
                *(uint4*)&Ks[r * 64 + ch * 8] = tk[i];
            }
        }
#endif
        // stage V tile transposed: Vts[dh][key] (batch loads, then LDS writes)
        {
            uint4 tv[2];
#pragma unroll
            for (int i = 0; i < 2; ++i) {
                int c = tid + i * 256, r = c >> 3, ch = c & 7;
                tv[i] = *(const uint4*)&Vb[(rowBase + kt + r) * CD + h * 64 + ch * 8];
            }
#pragma unroll
            for (int i = 0; i < 2; ++i) {
                int c = tid + i * 256, r = c >> 3, ch = c & 7;
                const bf16_t* p = (const bf16_t*)&tv[i];
#pragma unroll
                for (int j = 0; j < 8; ++j) Vts[(ch * 8 + j) * 64 + r] = p[j];
            }
        }
        async_wait_all();
        __syncthreads();

        // S = Q * K^T  (16 query rows x 64 keys per wave)
        v8f sacc[4];
#pragma unroll
        for (int t = 0; t < 4; ++t)
#pragma unroll
            for (int e = 0; e < 8; ++e) sacc[t][e] = 0.0f;

#pragma unroll
        for (int kk = 0; kk < 64; kk += 32) {
            Frag16 a, b[4];
            const int arow = w * 16 + l16;
            a.u[0] = *(const uint4*)&Qs[arow * 64 + kk + half * 8];
            a.u[1] = *(const uint4*)&Qs[arow * 64 + kk + half * 8 + 16];
#pragma unroll
            for (int t = 0; t < 4; ++t) {
                const int brow = t * 16 + l16;
                b[t].u[0] = *(const uint4*)&Ks[brow * 64 + kk + half * 8];
                b[t].u[1] = *(const uint4*)&Ks[brow * 64 + kk + half * 8 + 16];
            }
#pragma unroll
            for (int t = 0; t < 4; ++t)
                sacc[t] = __builtin_amdgcn_wmma_f32_16x16x32_bf16(
                    false, a.v, false, b[t].v, (short)0, sacc[t], false, false);
        }

        // online softmax (row stats per VGPR slot v; row = v + 8*half)
#pragma unroll
        for (int v = 0; v < 8; ++v) {
            const int qg = qBase + w * 16 + v + half * 8;
            float mx = -1.0e30f;
#pragma unroll
            for (int t = 0; t < 4; ++t) {
                float s = sacc[t][v] * scale;
                if (causal) {
                    const int kg = kt + t * 16 + l16;
                    s = (kg <= qg) ? s : -1.0e30f;
                }
                sacc[t][v] = s;
                mx = fmaxf(mx, s);
            }
            // reduce over the 16 lanes of this half (xor masks < 16 stay in half)
#pragma unroll
            for (int d = 1; d < 16; d <<= 1) mx = fmaxf(mx, __shfl_xor(mx, d, 32));
            const float mold = mrow[v];
            const float mnew = fmaxf(mold, mx);
            const float alpha = __expf(mold - mnew);
            mrow[v] = mnew;
            float rsum = 0.0f;
#pragma unroll
            for (int t = 0; t < 4; ++t) {
                float p = __expf(sacc[t][v] - mnew);
                sacc[t][v] = p;
                rsum += p;
            }
#pragma unroll
            for (int d = 1; d < 16; d <<= 1) rsum += __shfl_xor(rsum, d, 32);
            lrow[v] = lrow[v] * alpha + rsum;
#pragma unroll
            for (int t = 0; t < 4; ++t) oacc[t][v] *= alpha;
            // spill P (bf16) into this wave's private LDS strip (C->A relayout)
            const int prow = v + half * 8;
#pragma unroll
            for (int t = 0; t < 4; ++t)
                Ps[(w * 16 + prow) * 64 + t * 16 + l16] = (bf16_t)sacc[t][v];
        }
        // same-wave LDS ops are in-order: Ps writes land before reads below

        // O += P * V   (A = P [16 x 64keys], B = Vts [dh][key])
#pragma unroll
        for (int kk = 0; kk < 64; kk += 32) {
            Frag16 a, b[4];
            a.u[0] = *(const uint4*)&Ps[(w * 16 + l16) * 64 + kk + half * 8];
            a.u[1] = *(const uint4*)&Ps[(w * 16 + l16) * 64 + kk + half * 8 + 16];
#pragma unroll
            for (int t = 0; t < 4; ++t) {
                const int brow = t * 16 + l16;
                b[t].u[0] = *(const uint4*)&Vts[brow * 64 + kk + half * 8];
                b[t].u[1] = *(const uint4*)&Vts[brow * 64 + kk + half * 8 + 16];
            }
#pragma unroll
            for (int t = 0; t < 4; ++t)
                oacc[t] = __builtin_amdgcn_wmma_f32_16x16x32_bf16(
                    false, a.v, false, b[t].v, (short)0, oacc[t], false, false);
        }
    }

    // normalize and store (bf16, feeds the output-projection GEMM)
#pragma unroll
    for (int v = 0; v < 8; ++v) {
        const float inv = 1.0f / lrow[v];
        const size_t row = rowBase + qBase + w * 16 + v + half * 8;
#pragma unroll
        for (int t = 0; t < 4; ++t) {
            const int col = h * 64 + t * 16 + l16;
            Ob[row * CD + col] = (bf16_t)(oacc[t][v] * inv);
        }
    }
}

// ---------------------------------------------------------------------------
// out = LayerNorm(X + R) with unbiased variance (ddof=1), eps=1e-12.
// One block (256 threads) per row of D=1024. Emits fp32 and optional bf16.
// ---------------------------------------------------------------------------
__global__ __launch_bounds__(256) void residual_layernorm(const float* __restrict__ X,
                                                          const float* __restrict__ R,
                                                          const float* __restrict__ gam,
                                                          const float* __restrict__ bet,
                                                          float* __restrict__ outF,
                                                          bf16_t* __restrict__ outB) {
    __shared__ float red[16];
    const int row = blockIdx.x;
    const float* x = X + (size_t)row * CD;
    const float* r = R + (size_t)row * CD;

    float vals[4];
    float s = 0.0f, s2 = 0.0f;
#pragma unroll
    for (int i = 0; i < 4; ++i) {
        float v = x[threadIdx.x + i * 256] + r[threadIdx.x + i * 256];
        vals[i] = v;
        s += v;
        s2 += v * v;
    }
#pragma unroll
    for (int d = 1; d < 32; d <<= 1) {
        s  += __shfl_xor(s, d, 32);
        s2 += __shfl_xor(s2, d, 32);
    }
    const int w = threadIdx.x >> 5, lane = threadIdx.x & 31;
    if (lane == 0) { red[w] = s; red[8 + w] = s2; }
    __syncthreads();
    float ts = 0.0f, ts2 = 0.0f;
#pragma unroll
    for (int i = 0; i < 8; ++i) { ts += red[i]; ts2 += red[8 + i]; }
    const float mean = ts / (float)CD;
    const float var  = (ts2 - ts * mean) / (float)(CD - 1);
    const float inv  = rsqrtf(var + 1e-12f);
#pragma unroll
    for (int i = 0; i < 4; ++i) {
        const int c = threadIdx.x + i * 256;
        float y = gam[c] * (vals[i] - mean) * inv + bet[c];
        outF[(size_t)row * CD + c] = y;
        if (outB) outB[(size_t)row * CD + c] = (bf16_t)y;
    }
}

// ---------------------------------------------------------------------------
// Host side
// ---------------------------------------------------------------------------
extern "C" void kernel_launch(void* const* d_in, const int* in_sizes, int n_in,
                              void* d_out, int out_size, void* d_ws, size_t ws_size,
                              hipStream_t stream) {
    const float* dec   = (const float*)d_in[0];
    const float* enc   = (const float*)d_in[1];
    // d_in[2] (mask) == tril — applied analytically in the attention kernel.
    const float* sa_wq = (const float*)d_in[3];
    const float* sa_wk = (const float*)d_in[4];
    const float* sa_wv = (const float*)d_in[5];
    const float* sa_bq = (const float*)d_in[6];
    const float* sa_bk = (const float*)d_in[7];
    const float* sa_bv = (const float*)d_in[8];
    const float* sa_wo = (const float*)d_in[9];
    const float* sa_bo = (const float*)d_in[10];
    const float* ca_wq = (const float*)d_in[11];
    const float* ca_wk = (const float*)d_in[12];
    const float* ca_wv = (const float*)d_in[13];
    const float* ca_bq = (const float*)d_in[14];
    const float* ca_bk = (const float*)d_in[15];
    const float* ca_bv = (const float*)d_in[16];
    const float* ca_wo = (const float*)d_in[17];
    const float* ca_bo = (const float*)d_in[18];
    const float* ff_w1 = (const float*)d_in[19];
    const float* ff_b1 = (const float*)d_in[20];
    const float* ff_w2 = (const float*)d_in[21];
    const float* ff_b2 = (const float*)d_in[22];
    const float* ln1g = (const float*)d_in[23];
    const float* ln1b = (const float*)d_in[24];
    const float* ln2g = (const float*)d_in[25];
    const float* ln2b = (const float*)d_in[26];
    const float* ln3g = (const float*)d_in[27];
    const float* ln3b = (const float*)d_in[28];

    char* ws = (char*)d_ws;
    const size_t szDxD = (size_t)CD * CD * sizeof(bf16_t);   // 2 MB
    const size_t szDxF = (size_t)CD * CF * sizeof(bf16_t);   // 8 MB
    const size_t szND2 = (size_t)CN * CD * sizeof(bf16_t);   // 8 MB
    const size_t szND4 = (size_t)CN * CD * sizeof(float);    // 16 MB
    const size_t szNF2 = (size_t)CN * CF * sizeof(bf16_t);   // 32 MB

    size_t off = 0;
    bf16_t* wSAq = (bf16_t*)(ws + off); off += szDxD;
    bf16_t* wSAk = (bf16_t*)(ws + off); off += szDxD;
    bf16_t* wSAv = (bf16_t*)(ws + off); off += szDxD;
    bf16_t* wSAo = (bf16_t*)(ws + off); off += szDxD;
    bf16_t* wCAq = (bf16_t*)(ws + off); off += szDxD;
    bf16_t* wCAk = (bf16_t*)(ws + off); off += szDxD;
    bf16_t* wCAv = (bf16_t*)(ws + off); off += szDxD;
    bf16_t* wCAo = (bf16_t*)(ws + off); off += szDxD;
    bf16_t* wF1  = (bf16_t*)(ws + off); off += szDxF;
    bf16_t* wF2  = (bf16_t*)(ws + off); off += szDxF;
    bf16_t* Xb   = (bf16_t*)(ws + off); off += szND2;
    bf16_t* Eb   = (bf16_t*)(ws + off); off += szND2;
    bf16_t* Qb   = (bf16_t*)(ws + off); off += szND2;
    bf16_t* Kb   = (bf16_t*)(ws + off); off += szND2;
    bf16_t* Vb   = (bf16_t*)(ws + off); off += szND2;
    bf16_t* Ab   = (bf16_t*)(ws + off); off += szND2;
    float*  Sf   = (float*) (ws + off); off += szND4;
    float*  o1f  = (float*) (ws + off); off += szND4;
    bf16_t* o1b  = (bf16_t*)(ws + off); off += szND2;
    float*  o2f  = (float*) (ws + off); off += szND4;
    bf16_t* o2b  = (bf16_t*)(ws + off); off += szND2;
    bf16_t* Hb   = (bf16_t*)(ws + off); off += szNF2;
    (void)ws_size; (void)in_sizes; (void)n_in; (void)out_size;

    // ---- pack weights (transpose + bf16) ----
    const int P256 = 256;
    auto blocksFor = [](long long n) { return (unsigned)((n + 255) / 256); };
    const long long nQKV = (long long)CH * CD * CDH;   // 1M
    pack_transpose<<<blocksFor(nQKV), P256, 0, stream>>>(sa_wq, wSAq, CH, CD, CDH);
    pack_transpose<<<blocksFor(nQKV), P256, 0, stream>>>(sa_wk, wSAk, CH, CD, CDH);
    pack_transpose<<<blocksFor(nQKV), P256, 0, stream>>>(sa_wv, wSAv, CH, CD, CDH);
    pack_transpose<<<blocksFor((long long)CD * CD), P256, 0, stream>>>(sa_wo, wSAo, 1, CD, CD);
    pack_transpose<<<blocksFor(nQKV), P256, 0, stream>>>(ca_wq, wCAq, CH, CD, CDH);
    pack_transpose<<<blocksFor(nQKV), P256, 0, stream>>>(ca_wk, wCAk, CH, CD, CDH);
    pack_transpose<<<blocksFor(nQKV), P256, 0, stream>>>(ca_wv, wCAv, CH, CD, CDH);
    pack_transpose<<<blocksFor((long long)CD * CD), P256, 0, stream>>>(ca_wo, wCAo, 1, CD, CD);
    pack_transpose<<<blocksFor((long long)CD * CF), P256, 0, stream>>>(ff_w1, wF1, 1, CD, CF);
    pack_transpose<<<blocksFor((long long)CF * CD), P256, 0, stream>>>(ff_w2, wF2, 1, CF, CD);

    // ---- activations to bf16 ----
    const int nd4 = CN * CD / 4;
    cvt_f32_bf16<<<blocksFor(nd4), P256, 0, stream>>>(dec, Xb, nd4);
    cvt_f32_bf16<<<blocksFor(nd4), P256, 0, stream>>>(enc, Eb, nd4);

    const dim3 gD(CN / GM, CD / GN);   // (32,16) for D-wide outputs
    const dim3 gF(CN / GM, CF / GN);   // (32,64) for F-wide outputs
    const dim3 gA(CS / 128, CB * CH);  // attention: (8, 64)

    // ---- self-attention ----
    gemm_bf16<<<gD, P256, 0, stream>>>(Xb, wSAq, sa_bq, nullptr, Qb, CN, CD, CD, 0);
    gemm_bf16<<<gD, P256, 0, stream>>>(Xb, wSAk, sa_bk, nullptr, Kb, CN, CD, CD, 0);
    gemm_bf16<<<gD, P256, 0, stream>>>(Xb, wSAv, sa_bv, nullptr, Vb, CN, CD, CD, 0);
    attention_kernel<<<gA, P256, 0, stream>>>(Qb, Kb, Vb, Ab, /*causal=*/1);
    gemm_bf16<<<gD, P256, 0, stream>>>(Ab, wSAo, sa_bo, Sf, nullptr, CN, CD, CD, 0);
    residual_layernorm<<<CN, P256, 0, stream>>>(Sf, dec, ln1g, ln1b, o1f, o1b);

    // ---- cross-attention ----
    gemm_bf16<<<gD, P256, 0, stream>>>(o1b, wCAq, ca_bq, nullptr, Qb, CN, CD, CD, 0);
    gemm_bf16<<<gD, P256, 0, stream>>>(Eb,  wCAk, ca_bk, nullptr, Kb, CN, CD, CD, 0);
    gemm_bf16<<<gD, P256, 0, stream>>>(Eb,  wCAv, ca_bv, nullptr, Vb, CN, CD, CD, 0);
    attention_kernel<<<gA, P256, 0, stream>>>(Qb, Kb, Vb, Ab, /*causal=*/0);
    gemm_bf16<<<gD, P256, 0, stream>>>(Ab, wCAo, ca_bo, Sf, nullptr, CN, CD, CD, 0);
    residual_layernorm<<<CN, P256, 0, stream>>>(Sf, o1f, ln2g, ln2b, o2f, o2b);

    // ---- feed-forward ----
    gemm_bf16<<<gF, P256, 0, stream>>>(o2b, wF1, ff_b1, nullptr, Hb, CN, CF, CD, /*relu=*/1);
    gemm_bf16<<<gD, P256, 0, stream>>>(Hb,  wF2, ff_b2, Sf, nullptr, CN, CD, CF, 0);
    residual_layernorm<<<CN, P256, 0, stream>>>(Sf, o2f, ln3g, ln3b, (float*)d_out, nullptr);
}
